// LLaMABaseline_24936580120935
// MI455X (gfx1250) — matrix operand order repaired and checked by reading, not compiled, not run on hardware
//
#include <hip/hip_runtime.h>
#include <hip/hip_bf16.h>
#include <stdint.h>

// ---- problem dims (compile-time) ----
constexpr int L_ = 8, D_ = 1024, H_ = 16, DH_ = 64, DI_ = 4096, V_ = 32000, B_ = 2, S_ = 1024;
constexpr int M_ = B_ * S_;   // 2048 token rows

typedef __bf16 bf16_t;
typedef __attribute__((ext_vector_type(16))) __bf16 v16bf;
typedef __attribute__((ext_vector_type(8)))  __bf16 v8bf;
typedef __attribute__((ext_vector_type(4)))  __bf16 v4bf;
typedef __attribute__((ext_vector_type(8)))  float  v8f;
typedef __attribute__((ext_vector_type(4)))  unsigned int u32x4;
typedef __attribute__((ext_vector_type(8)))  int          i32x8;
typedef __attribute__((ext_vector_type(4)))  int          i32x4;

__device__ inline v16bf cat8(v8bf lo, v8bf hi) {
  return __builtin_shufflevector(lo, hi, 0,1,2,3,4,5,6,7,8,9,10,11,12,13,14,15);
}
__device__ inline v8f vzero8() { v8f z = {0.f,0.f,0.f,0.f,0.f,0.f,0.f,0.f}; return z; }

#define WMMA_BF16(a,b,c) __builtin_amdgcn_wmma_f32_16x16x32_bf16(false,(a),false,(b),(short)0,(c),false,false)

// ---- TDM: 2D tile (tileRows x tileK bf16 elems, row stride strideElems) -> LDS ----
// D# bit layout per CDNA5 ISA ch.8 (group0: count/lds/global/type; group1: dims/strides)
// amdgpu-toolchain clang-23 builtin: 6 args (g0 u32x4, g1 i32x8, g2 i32x4, g3 i32x4, i32x8, cpol)
__device__ inline void tdm_load_2d(unsigned lds_off, const void* gaddr,
                                   unsigned tileK, unsigned tileRows, unsigned strideElems) {
  unsigned long long ga = (unsigned long long)(uintptr_t)gaddr;
  u32x4 g0;
  g0[0] = 1u;                                            // count=1, user descriptor
  g0[1] = lds_off;                                       // lds_addr (bytes)
  g0[2] = (unsigned)ga;                                  // global_addr[31:0]
  g0[3] = (unsigned)((ga >> 32) & 0x1FFFFFFu) | (2u << 30); // global_addr[56:32], type=2
  unsigned td0 = strideElems;                            // tensor_dim0 (row length)
  unsigned td1 = 0x7FFFFFu;                              // tensor_dim1 (ample rows)
  i32x8 g1;
  g1[0] = (int)(1u << 16);                               // data_size = 2 bytes
  g1[1] = (int)((td0 & 0xFFFFu) << 16);                  // tensor_dim0[15:0]
  g1[2] = (int)((td0 >> 16) | ((td1 & 0xFFFFu) << 16));  // td0[31:16] | td1[15:0]
  g1[3] = (int)((td1 >> 16) | (tileK << 16));            // td1[31:16] | tile_dim0
  g1[4] = (int)tileRows;                                 // tile_dim1 (tile_dim2 = 0)
  g1[5] = (int)strideElems;                              // tensor_dim0_stride[31:0]
  g1[6] = 0;                                             // stride[47:32], dim1_stride lo
  g1[7] = 0;
  i32x4 z4 = {0, 0, 0, 0};
  i32x8 z8 = {0, 0, 0, 0, 0, 0, 0, 0};
  __builtin_amdgcn_tensor_load_to_lds(g0, g1, z4, z4, z8, 0);
}

// ---------------- fp32 -> bf16 bulk convert ----------------
__global__ void k_cvt_bf16(const float* __restrict__ in, bf16_t* __restrict__ out, size_t n4) {
  size_t i = (size_t)blockIdx.x * blockDim.x + threadIdx.x;
  size_t stride = (size_t)gridDim.x * blockDim.x;
  for (; i < n4; i += stride) {
    float4 f = reinterpret_cast<const float4*>(in)[i];
    v4bf o; o[0] = (bf16_t)f.x; o[1] = (bf16_t)f.y; o[2] = (bf16_t)f.z; o[3] = (bf16_t)f.w;
    reinterpret_cast<v4bf*>(out)[i] = o;
  }
}

// ---------------- embedding gather (fp32 residual stream) ----------------
__global__ void k_embed(const int* __restrict__ x, const float* __restrict__ emb,
                        float* __restrict__ h) {
  int idx = blockIdx.x * blockDim.x + threadIdx.x;   // M_ * (D_/4)
  int m  = idx >> 8;                                  // D_/4 = 256 float4 per row
  int d4 = idx & 255;
  float4 f = reinterpret_cast<const float4*>(emb + (size_t)x[m] * D_)[d4];
  reinterpret_cast<float4*>(h + (size_t)m * D_)[d4] = f;
}

// ---------------- RMSNorm (fp32 in, bf16 out) ----------------
__global__ __launch_bounds__(256) void k_rmsnorm(const float* __restrict__ x,
                                                 const float* __restrict__ w,
                                                 bf16_t* __restrict__ out) {
  int r = blockIdx.x, t = threadIdx.x;
  float4 f = reinterpret_cast<const float4*>(x + (size_t)r * D_)[t];
  float ss = f.x*f.x + f.y*f.y + f.z*f.z + f.w*f.w;
  #pragma unroll
  for (int o = 16; o; o >>= 1) ss += __shfl_xor(ss, o, 32);
  __shared__ float wred[8];
  if ((t & 31) == 0) wred[t >> 5] = ss;
  __syncthreads();
  float tot = 0.f;
  #pragma unroll
  for (int i = 0; i < 8; ++i) tot += wred[i];
  float inv = rsqrtf(tot * (1.0f / D_) + 1e-6f);
  float4 wv = reinterpret_cast<const float4*>(w)[t];
  v4bf o;
  o[0] = (bf16_t)(f.x * wv.x * inv); o[1] = (bf16_t)(f.y * wv.y * inv);
  o[2] = (bf16_t)(f.z * wv.z * inv); o[3] = (bf16_t)(f.w * wv.w * inv);
  reinterpret_cast<v4bf*>(out + (size_t)r * D_)[t] = o;
}

// ---------------- TDM-fed, LDS double-buffered WMMA GEMM ----------------
// out[M,N] = A[M,K] . W[N,K]^T (+bias)(+res)
// block = 256 thr = 8 waves (4m x 2n); block tile 128x128; wave tile 32x64 (2x4 WMMA)
constexpr int BM_ = 128, BN_ = 128, KC_ = 64;
constexpr size_t GEMM_LDS = (size_t)2 * (BM_ * KC_ + BN_ * KC_) * 2;   // 64 KB

__global__ __launch_bounds__(256) void k_gemm_tdm(
    const bf16_t* __restrict__ A, const bf16_t* __restrict__ W,
    const float* __restrict__ bias, const float* __restrict__ res,
    float* __restrict__ out, int N, int K) {
  extern __shared__ char smem[];
  bf16_t* lA = (bf16_t*)smem;                              // [2][BM_*KC_]
  bf16_t* lB = (bf16_t*)(smem + (size_t)2 * BM_ * KC_ * 2);// [2][BN_*KC_]
  int lane = threadIdx.x & 31;
  int wave = threadIdx.x >> 5;
  int wm = wave >> 1, wn = wave & 1;
  int ln   = lane & 15;
  int aoff = lane & 16;          // 0 or 16
  int boff = lane & 16;          // 0 or 16
  int hh   = (lane & 16) >> 1;
  int blockM = blockIdx.y * BM_;
  int blockN = blockIdx.x * BN_;
  int nchunks = K / KC_;

  auto issue = [&](int c) {
    const bf16_t* gA = A + (size_t)blockM * K + c * KC_;
    const bf16_t* gB = W + (size_t)blockN * K + c * KC_;
    tdm_load_2d((unsigned)(uintptr_t)(lA + (c & 1) * (BM_ * KC_)), gA, KC_, BM_, (unsigned)K);
    tdm_load_2d((unsigned)(uintptr_t)(lB + (c & 1) * (BN_ * KC_)), gB, KC_, BN_, (unsigned)K);
  };

  if (wave == 0) issue(0);

  v8f acc[2][4];
  #pragma unroll
  for (int sm = 0; sm < 2; ++sm)
    #pragma unroll
    for (int sn = 0; sn < 4; ++sn) acc[sm][sn] = vzero8();

  for (int c = 0; c < nchunks; ++c) {
    if (wave == 0) {
      if (c + 1 < nchunks) { issue(c + 1); __builtin_amdgcn_s_wait_tensorcnt(2); }
      else                 { __builtin_amdgcn_s_wait_tensorcnt(0); }
    }
    __syncthreads();                       // chunk c resident in LDS
    const bf16_t* bufA = lA + (c & 1) * (BM_ * KC_);
    const bf16_t* bufB = lB + (c & 1) * (BN_ * KC_);
    #pragma unroll
    for (int kk = 0; kk < KC_; kk += 32) {
      v16bf af[2], bfr[4];
      #pragma unroll
      for (int sm = 0; sm < 2; ++sm) {
        const bf16_t* ar = bufA + (size_t)(32*wm + 16*sm + ln) * KC_ + kk + aoff;
        af[sm] = cat8(*reinterpret_cast<const v8bf*>(ar),
                      *reinterpret_cast<const v8bf*>(ar + 8));
      }
      #pragma unroll
      for (int sn = 0; sn < 4; ++sn) {
        const bf16_t* br = bufB + (size_t)(64*wn + 16*sn + ln) * KC_ + kk + boff;
        bfr[sn] = *reinterpret_cast<const v16bf*>(br);
      }
      #pragma unroll
      for (int sm = 0; sm < 2; ++sm)
        #pragma unroll
        for (int sn = 0; sn < 4; ++sn)
          acc[sm][sn] = WMMA_BF16(af[sm], bfr[sn], acc[sm][sn]);
    }
    __syncthreads();                       // all reads done before buffer re-fill
  }

  #pragma unroll
  for (int sm = 0; sm < 2; ++sm) {
    #pragma unroll
    for (int sn = 0; sn < 4; ++sn) {
      int mb = blockM + 32*wm + 16*sm;
      int nb = blockN + 64*wn + 16*sn + ln;
      float bv_ = bias ? bias[nb] : 0.f;
      #pragma unroll
      for (int v = 0; v < 8; ++v) {
        size_t o = (size_t)(mb + v + hh) * N + nb;
        out[o] = acc[sm][sn][v] + bv_ + (res ? res[o] : 0.f);
      }
    }
  }
}

// ---------------- RoPE + head-major relayout; V transposed to [B,H,DH,S] ----------------
__global__ void k_rope(const float* __restrict__ qf, const float* __restrict__ kf,
                       const float* __restrict__ vf, bf16_t* __restrict__ qb,
                       bf16_t* __restrict__ kb, bf16_t* __restrict__ vtb) {
  int idx = blockIdx.x * blockDim.x + threadIdx.x;   // B*S*H*DH
  int d  = idx & 63;
  int hd = (idx >> 6) & 15;
  int s  = (idx >> 10) & 1023;
  int b  = idx >> 20;
  size_t src = (size_t)(b * S_ + s) * D_ + hd * DH_ + d;
  int i = d & 31;
  float inv_freq = exp2f(-(float)i * (13.287712379549449f / 32.0f));  // 10000^(-i/32)
  float sn, cs;
  sincosf((float)s * inv_freq, &sn, &cs);
  size_t srot = (d < 32) ? src + 32 : src - 32;
  float sgn = (d < 32) ? -1.f : 1.f;
  float q0 = qf[src], k0 = kf[src];
  float qr = sgn * qf[srot], kr = sgn * kf[srot];
  size_t dq = ((size_t)(b * H_ + hd) * S_ + s) * DH_ + d;
  qb[dq] = (bf16_t)(q0 * cs + qr * sn);
  kb[dq] = (bf16_t)(k0 * cs + kr * sn);
  vtb[((size_t)(b * H_ + hd) * DH_ + d) * S_ + s] = (bf16_t)vf[src];
}

// ---------------- SwiGLU: bf16(silu(g) * u) ----------------
__global__ void k_swiglu(const float* __restrict__ g, const float* __restrict__ u,
                         bf16_t* __restrict__ out, int n) {
  int i = blockIdx.x * blockDim.x + threadIdx.x;
  if (i < n) {
    float gv = g[i];
    out[i] = (bf16_t)((gv / (1.f + __expf(-gv))) * u[i]);
  }
}

// ---------------- Flash attention: 64-query tile per WG, online softmax ----------------
__global__ __launch_bounds__(256) void k_attn(const bf16_t* __restrict__ qb,
                                              const bf16_t* __restrict__ kb,
                                              const bf16_t* __restrict__ vtb,
                                              bf16_t* __restrict__ yb) {
  __shared__ float  s_s[64 * 66];   // padded score tile
  __shared__ bf16_t s_p[64 * 72];   // padded P tile (bf16)
  __shared__ float  s_m[64], s_l[64], s_a[64];
  const float scale = 0.125f;                 // 1/sqrt(64)
  const float L2E = 1.4426950408889634f;
  int tid = threadIdx.x, lane = tid & 31, wave = tid >> 5;
  int mi = wave >> 1;
  int ni0 = (wave & 1) * 2;
  int ln = lane & 15;
  int hi = lane & 16;
  int hh = hi >> 1;
  int bh = blockIdx.y;
  int qt = blockIdx.x;
  int qbase = qt * 64;
  const bf16_t* qp = qb  + (size_t)bh * S_ * DH_;
  const bf16_t* kp = kb  + (size_t)bh * S_ * DH_;
  const bf16_t* vp = vtb + (size_t)bh * DH_ * S_;

  v16bf qa[2];
  {
    const bf16_t* qr = qp + (size_t)(qbase + 16*mi + ln) * DH_;
    #pragma unroll
    for (int ks = 0; ks < 2; ++ks) {
      int off = ks * 32 + hi;
      qa[ks] = cat8(*reinterpret_cast<const v8bf*>(qr + off),
                    *reinterpret_cast<const v8bf*>(qr + off + 8));
    }
  }
  v8f oacc[2] = {vzero8(), vzero8()};
  if (tid < 64) { s_m[tid] = -1e30f; s_l[tid] = 0.f; }
  __syncthreads();

  for (int kt = 0; kt <= qt; ++kt) {
    int kbase = kt * 64;
    v8f sacc[2] = {vzero8(), vzero8()};
    #pragma unroll
    for (int ks = 0; ks < 2; ++ks) {
      #pragma unroll
      for (int j = 0; j < 2; ++j) {
        const bf16_t* kr = kp + (size_t)(kbase + 16*(ni0+j) + ln) * DH_ + ks*32 + hi;
        sacc[j] = WMMA_BF16(qa[ks], *reinterpret_cast<const v16bf*>(kr), sacc[j]);
      }
    }
    #pragma unroll
    for (int j = 0; j < 2; ++j) {
      int ncol = 16*(ni0+j) + ln;
      int key = kbase + ncol;
      #pragma unroll
      for (int v = 0; v < 8; ++v) {
        int row = 16*mi + v + hh;
        s_s[row*66 + ncol] = (key <= qbase + row) ? sacc[j][v] * scale : -1e30f;
      }
    }
    __syncthreads();
    if (tid < 64) {
      int r = tid;
      float mo = s_m[r], mx = mo;
      const float* sr = &s_s[r*66];
      #pragma unroll 8
      for (int c = 0; c < 64; ++c) mx = fmaxf(mx, sr[c]);
      float sum = 0.f;
      bf16_t* pr = &s_p[r*72];
      #pragma unroll 8
      for (int c = 0; c < 64; ++c) {
        float pv = exp2f((sr[c] - mx) * L2E);
        sum += pv;
        pr[c] = (bf16_t)pv;
      }
      float al = exp2f((mo - mx) * L2E);
      s_a[r] = al; s_m[r] = mx;
      s_l[r] = s_l[r] * al + sum;
    }
    __syncthreads();
    float av[8];
    #pragma unroll
    for (int v = 0; v < 8; ++v) av[v] = s_a[16*mi + v + hh];
    #pragma unroll
    for (int j = 0; j < 2; ++j)
      #pragma unroll
      for (int v = 0; v < 8; ++v) oacc[j][v] *= av[v];
    #pragma unroll
    for (int ks = 0; ks < 2; ++ks) {
      const bf16_t* pr = &s_p[(16*mi + ln) * 72];
      int off = ks * 32 + hi;
      v16bf pa = cat8(*reinterpret_cast<const v8bf*>(pr + off),
                      *reinterpret_cast<const v8bf*>(pr + off + 8));
      #pragma unroll
      for (int j = 0; j < 2; ++j) {
        const bf16_t* vr = vp + (size_t)(16*(ni0+j) + ln) * S_ + kbase + ks*32 + hi;
        oacc[j] = WMMA_BF16(pa, *reinterpret_cast<const v16bf*>(vr), oacc[j]);
      }
    }
  }
  float rl[8];
  #pragma unroll
  for (int v = 0; v < 8; ++v) rl[v] = 1.f / s_l[16*mi + v + hh];
  int b = bh >> 4, hd = bh & 15;   // H_ == 16
  #pragma unroll
  for (int j = 0; j < 2; ++j) {
    int dh = 16*(ni0+j) + ln;
    #pragma unroll
    for (int v = 0; v < 8; ++v) {
      int row = 16*mi + v + hh;
      yb[(size_t)(b*S_ + qbase + row) * D_ + hd*DH_ + dh] = (bf16_t)(oacc[j][v] * rl[v]);
    }
  }
}

// ---------------- host orchestration ----------------
extern "C" void kernel_launch(void* const* d_in, const int* in_sizes, int n_in,
                              void* d_out, int out_size, void* d_ws, size_t ws_size,
                              hipStream_t stream) {
  (void)in_sizes; (void)n_in; (void)out_size; (void)ws_size;
  const int*   x   = (const int*)d_in[0];
  const float* emb = (const float*)d_in[1];
  const float* wq  = (const float*)d_in[2];
  const float* bq  = (const float*)d_in[3];
  const float* wk  = (const float*)d_in[4];
  const float* bk  = (const float*)d_in[5];
  const float* wv  = (const float*)d_in[6];
  const float* bv  = (const float*)d_in[7];
  const float* wo  = (const float*)d_in[8];
  const float* bo  = (const float*)d_in[9];
  const float* wg  = (const float*)d_in[10];
  const float* wu  = (const float*)d_in[11];
  const float* wd  = (const float*)d_in[12];
  const float* ln1 = (const float*)d_in[13];
  const float* ln2 = (const float*)d_in[14];
  const float* nw  = (const float*)d_in[15];
  float* logits = (float*)d_out;

  char* p = (char*)d_ws;
  auto alloc = [&](size_t bytes) { char* r = p; p += (bytes + 255) & ~(size_t)255; return r; };
  const size_t eDD = (size_t)L_ * D_ * D_, eID = (size_t)L_ * DI_ * D_;
  bf16_t* emb_b = (bf16_t*)alloc((size_t)V_ * D_ * 2);
  bf16_t* wq_b  = (bf16_t*)alloc(eDD * 2);
  bf16_t* wk_b  = (bf16_t*)alloc(eDD * 2);
  bf16_t* wv_b  = (bf16_t*)alloc(eDD * 2);
  bf16_t* wo_b  = (bf16_t*)alloc(eDD * 2);
  bf16_t* wg_b  = (bf16_t*)alloc(eID * 2);
  bf16_t* wu_b  = (bf16_t*)alloc(eID * 2);
  bf16_t* wd_b  = (bf16_t*)alloc(eID * 2);
  float*  h     = (float*)alloc((size_t)M_ * D_ * 4);
  bf16_t* hn_b  = (bf16_t*)alloc((size_t)M_ * D_ * 2);
  float*  qf    = (float*)alloc((size_t)M_ * D_ * 4);
  float*  kf    = (float*)alloc((size_t)M_ * D_ * 4);
  float*  vf    = (float*)alloc((size_t)M_ * D_ * 4);
  bf16_t* q_b   = (bf16_t*)alloc((size_t)M_ * D_ * 2);
  bf16_t* k_b   = (bf16_t*)alloc((size_t)M_ * D_ * 2);
  bf16_t* vt_b  = (bf16_t*)alloc((size_t)M_ * D_ * 2);
  bf16_t* y_b   = (bf16_t*)alloc((size_t)M_ * D_ * 2);
  float*  gatef = (float*)alloc((size_t)M_ * DI_ * 4);
  float*  upf   = (float*)alloc((size_t)M_ * DI_ * 4);
  bf16_t* act_b = (bf16_t*)alloc((size_t)M_ * DI_ * 2);

  auto cvt = [&](const float* in, bf16_t* outp, size_t n) {
    size_t n4 = n / 4;
    int blocks = (int)((n4 + 255) / 256);
    k_cvt_bf16<<<blocks, 256, 0, stream>>>(in, outp, n4);
  };
  cvt(emb, emb_b, (size_t)V_ * D_);
  cvt(wq, wq_b, eDD); cvt(wk, wk_b, eDD); cvt(wv, wv_b, eDD); cvt(wo, wo_b, eDD);
  cvt(wg, wg_b, eID); cvt(wu, wu_b, eID); cvt(wd, wd_b, eID);

  k_embed<<<M_, 256, 0, stream>>>(x, emb, h);

  dim3 gD(D_ / BN_, M_ / BM_);     // (8,16)
  dim3 gDI(DI_ / BN_, M_ / BM_);   // (32,16)
  dim3 gV(V_ / BN_, M_ / BM_);     // (250,16)
  const size_t shmem = GEMM_LDS;   // 64 KB dynamic LDS

  for (int l = 0; l < L_; ++l) {
    const size_t oDD = (size_t)l * D_ * D_, oID = (size_t)l * DI_ * D_, oD = (size_t)l * D_;
    // attention
    k_rmsnorm<<<M_, 256, 0, stream>>>(h, ln1 + oD, hn_b);
    k_gemm_tdm<<<gD, 256, shmem, stream>>>(hn_b, wq_b + oDD, bq + oD, nullptr, qf, D_, D_);
    k_gemm_tdm<<<gD, 256, shmem, stream>>>(hn_b, wk_b + oDD, bk + oD, nullptr, kf, D_, D_);
    k_gemm_tdm<<<gD, 256, shmem, stream>>>(hn_b, wv_b + oDD, bv + oD, nullptr, vf, D_, D_);
    k_rope<<<(M_ * D_) / 256, 256, 0, stream>>>(qf, kf, vf, q_b, k_b, vt_b);
    k_attn<<<dim3(S_ / 64, B_ * H_), 256, 0, stream>>>(q_b, k_b, vt_b, y_b);
    k_gemm_tdm<<<gD, 256, shmem, stream>>>(y_b, wo_b + oDD, bo + oD, h, h, D_, D_);
    // SwiGLU MLP
    k_rmsnorm<<<M_, 256, 0, stream>>>(h, ln2 + oD, hn_b);
    k_gemm_tdm<<<gDI, 256, shmem, stream>>>(hn_b, wg_b + oID, nullptr, nullptr, gatef, DI_, D_);
    k_gemm_tdm<<<gDI, 256, shmem, stream>>>(hn_b, wu_b + oID, nullptr, nullptr, upf, DI_, D_);
    k_swiglu<<<(M_ * DI_) / 256, 256, 0, stream>>>(gatef, upf, act_b, M_ * DI_);
    k_gemm_tdm<<<gD, 256, shmem, stream>>>(act_b, wd_b + oID, nullptr, h, h, D_, DI_);
  }

  k_rmsnorm<<<M_, 256, 0, stream>>>(h, nw, hn_b);
  k_gemm_tdm<<<gV, 256, shmem, stream>>>(hn_b, emb_b, nullptr, nullptr, logits, V_, D_);
}
